// Self_attention_28621662060791
// MI455X (gfx1250) — compile-verified, matrix operand order checked by reading, not run on hardware
//
#include <hip/hip_runtime.h>

typedef __bf16 bf16_t;
typedef __attribute__((ext_vector_type(16))) __bf16 v16bf;
typedef __attribute__((ext_vector_type(8)))  __bf16 v8bf;
typedef __attribute__((ext_vector_type(4)))  __bf16 v4bf;
typedef __attribute__((ext_vector_type(8)))  float  v8f;

#define E_DIM 1024
#define H_DIM 16
#define D_DIM 64
#define L_DIM 2048

// ---------------------------------------------------------------------------
// Fragment loaders for V_WMMA_F32_16X16X32_BF16 (wave32).
// 16-bit A (16x32, MxK) per-lane layout (ISA 7.12.2):
//   lanes 0-15 : row M=lane,    K = {0..7} in VGPR0-3, {16..23} in VGPR4-7
//   lanes 16-31: row M=lane-16, K = {8..15},           {24..31}
// B (32x16, KxN) mirrors this with lane <-> column N; since we always feed
// B as rows of a row-major matrix holding B^T (K contiguous), the same
// loader serves both A and B ("NT" GEMM).
// ---------------------------------------------------------------------------
__device__ __forceinline__ v16bf load_frag_nt(const bf16_t* __restrict__ base,
                                              int row0, int lda, int k0) {
  const int lane = threadIdx.x & 31;
  const int half = lane >> 4;
  const bf16_t* p = base + (size_t)(row0 + (lane & 15)) * lda + k0 + half * 8;
  v8bf c0 = *(const v8bf*)(p);
  v8bf c1 = *(const v8bf*)(p + 16);
  return __builtin_shufflevector(c0, c1, 0,1,2,3,4,5,6,7,8,9,10,11,12,13,14,15);
}

__device__ __forceinline__ v16bf load_frag_lds(const bf16_t* base) {
  const int lane = threadIdx.x & 31;
  const int half = lane >> 4;
  const bf16_t* p = base + (lane & 15) * 32 + half * 8;
  v8bf c0 = *(const v8bf*)(p);
  v8bf c1 = *(const v8bf*)(p + 16);
  return __builtin_shufflevector(c0, c1, 0,1,2,3,4,5,6,7,8,9,10,11,12,13,14,15);
}

__device__ __forceinline__ float half16_max(float v) {
  v = fmaxf(v, __shfl_xor(v, 1, 32));
  v = fmaxf(v, __shfl_xor(v, 2, 32));
  v = fmaxf(v, __shfl_xor(v, 4, 32));
  v = fmaxf(v, __shfl_xor(v, 8, 32));
  return v;
}
__device__ __forceinline__ float half16_sum(float v) {
  v += __shfl_xor(v, 1, 32);
  v += __shfl_xor(v, 2, 32);
  v += __shfl_xor(v, 4, 32);
  v += __shfl_xor(v, 8, 32);
  return v;
}

// ---------------------------------------------------------------------------
// f32 -> bf16 conversion (vectorized: float4 in, 4x bf16 out)
// ---------------------------------------------------------------------------
__global__ void cvt_kernel(const float* __restrict__ in, bf16_t* __restrict__ out, int n4) {
  int i = blockIdx.x * blockDim.x + threadIdx.x;
  if (i < n4) {
    float4 v = ((const float4*)in)[i];
    v4bf o;
    o[0] = (bf16_t)v.x; o[1] = (bf16_t)v.y; o[2] = (bf16_t)v.z; o[3] = (bf16_t)v.w;
    ((v4bf*)out)[i] = o;
  }
}

// ---------------------------------------------------------------------------
// NT GEMM: out[i][j] = sum_k A[i][k]*B[j][k]  (+ pos[i]*wabs[j]) (+ bias[j])
// block = 256 threads (8 waves), 128x128 tile, each wave 32x64 (2x4 WMMA C).
// Compile-time modes keep the epilogue branch-free:
//   OMODE 0: bf16 out[row*N+col]    OMODE 1: bf16 out[col*M+row] (transposed)
//   OMODE 2: f32  out[row*N+col]
// ---------------------------------------------------------------------------
template <int OMODE, bool POSB, bool BIAS>
__global__ __launch_bounds__(256)
void gemm_nt_kernel(const bf16_t* __restrict__ A, const bf16_t* __restrict__ B,
                    int M, int N, int K,
                    const float* __restrict__ pos, const float* __restrict__ wabs,
                    const float* __restrict__ bias,
                    bf16_t* __restrict__ outB, float* __restrict__ outF) {
  const int lane = threadIdx.x & 31;
  const int wave = threadIdx.x >> 5;
  const int wm = wave & 3, wn = wave >> 2;
  const int m0 = blockIdx.y * 128 + wm * 32;
  const int n0 = blockIdx.x * 128 + wn * 64;

  v8f zero = {};
  v8f acc[2][4];
#pragma unroll
  for (int i = 0; i < 2; i++)
#pragma unroll
    for (int t = 0; t < 4; t++) acc[i][t] = zero;

  for (int k0 = 0; k0 < K; k0 += 32) {
    v16bf a[2], b[4];
    a[0] = load_frag_nt(A, m0,      K, k0);
    a[1] = load_frag_nt(A, m0 + 16, K, k0);
#pragma unroll
    for (int t = 0; t < 4; t++) b[t] = load_frag_nt(B, n0 + t * 16, K, k0);
#pragma unroll
    for (int i = 0; i < 2; i++)
#pragma unroll
      for (int t = 0; t < 4; t++)
        acc[i][t] = __builtin_amdgcn_wmma_f32_16x16x32_bf16(
            false, a[i], false, b[t], (short)0, acc[i][t], false, false);
  }

  const int half = lane >> 4;
  const int cl = lane & 15;

  // Hoist per-row position values (uniform across t / cl).
  float pv[2][8];
  if (POSB) {
#pragma unroll
    for (int i = 0; i < 2; i++)
#pragma unroll
      for (int r = 0; r < 8; r++) pv[i][r] = pos[m0 + i * 16 + r + half * 8];
  }

#pragma unroll
  for (int i = 0; i < 2; i++) {
#pragma unroll
    for (int t = 0; t < 4; t++) {
      const int colg = n0 + t * 16 + cl;
      float addc = 0.f;
      if (BIAS) addc = bias[colg];
      float wab = 0.f;
      if (POSB) wab = wabs[colg];
#pragma unroll
      for (int r = 0; r < 8; r++) {
        const int rowg = m0 + i * 16 + r + half * 8;
        float v = acc[i][t][r] + addc;
        if (POSB) v += pv[i][r] * wab;
        if (OMODE == 2)      outF[(size_t)rowg * N + colg] = v;
        else if (OMODE == 1) outB[(size_t)colg * M + rowg] = (bf16_t)v;
        else                 outB[(size_t)rowg * N + colg] = (bf16_t)v;
      }
    }
  }
}

// ---------------------------------------------------------------------------
// qw[h, i] = sum_d q[i, h*64+d] * W_k_rel[h*64+d]
// ---------------------------------------------------------------------------
__global__ void qw_kernel(const bf16_t* __restrict__ qp, const float* __restrict__ wkrel,
                          float* __restrict__ qw) {
  int tid = blockIdx.x * blockDim.x + threadIdx.x;
  if (tid >= H_DIM * L_DIM) return;
  int i = tid & (L_DIM - 1);
  int h = tid >> 11;
  const bf16_t* qrow = qp + (size_t)i * E_DIM + h * D_DIM;
  const float* w = wkrel + h * D_DIM;
  float s = 0.f;
#pragma unroll 8
  for (int d = 0; d < D_DIM; d++) s += (float)qrow[d] * w[d];
  qw[h * L_DIM + i] = s;
}

// ---------------------------------------------------------------------------
// Flash attention with relative-position bias.
// One wave handles 16 query rows of one head; 8 independent waves per block.
// kp includes pos*W_k_abs, vpT is the V projection stored transposed [E, L].
// energy = (q.k + qw[h,q]*rel)/sqrt(E), rel = min(|pi-pj|,5000)/5000.
// ctx[q, h*64+d] = sum_j softmax * v  +  (sum_j softmax*rel) * wV[h,d]
// ---------------------------------------------------------------------------
__global__ __launch_bounds__(256)
void flash_kernel(const bf16_t* __restrict__ qp, const bf16_t* __restrict__ kp,
                  const bf16_t* __restrict__ vpT, const float* __restrict__ qw,
                  const float* __restrict__ position, const float* __restrict__ wvrel,
                  bf16_t* __restrict__ ctx) {
  __shared__ __align__(16) bf16_t Plds[8][16 * 32];
  const int lane = threadIdx.x & 31;
  const int wave = threadIdx.x >> 5;
  const int h = blockIdx.x & (H_DIM - 1);
  const int i0 = (blockIdx.x >> 4) * 128 + wave * 16;
  const int half = lane >> 4, cl = lane & 15;
  bf16_t* pl = &Plds[wave][0];

  // Q fragments for this 16-row tile (A 16x64 = two K=32 fragments), kept live.
  const v16bf qa0 = load_frag_nt(qp + h * D_DIM, i0, E_DIM, 0);
  const v16bf qa1 = load_frag_nt(qp + h * D_DIM, i0, E_DIM, 32);

  float mrow[8], lrow[8], srow[8], posr[8], qwv[8];
#pragma unroll
  for (int r = 0; r < 8; r++) {
    const int row = i0 + r + half * 8;
    mrow[r] = -1e30f; lrow[r] = 0.f; srow[r] = 0.f;
    posr[r] = position[row];
    qwv[r]  = qw[h * L_DIM + row];
  }
  v8f zero = {};
  v8f O[4];
#pragma unroll
  for (int t = 0; t < 4; t++) O[t] = zero;

  const float scale = 0.03125f;   // 1/sqrt(1024)
  for (int j0 = 0; j0 < L_DIM; j0 += 32) {
    // S = Q @ K^T for two 16-key sub-blocks
    v8f S0 = zero, S1 = zero;
    {
      v16bf kb0 = load_frag_nt(kp + h * D_DIM, j0,      E_DIM, 0);
      v16bf kb1 = load_frag_nt(kp + h * D_DIM, j0,      E_DIM, 32);
      S0 = __builtin_amdgcn_wmma_f32_16x16x32_bf16(false, qa0, false, kb0, (short)0, S0, false, false);
      S0 = __builtin_amdgcn_wmma_f32_16x16x32_bf16(false, qa1, false, kb1, (short)0, S0, false, false);
      v16bf kb2 = load_frag_nt(kp + h * D_DIM, j0 + 16, E_DIM, 0);
      v16bf kb3 = load_frag_nt(kp + h * D_DIM, j0 + 16, E_DIM, 32);
      S1 = __builtin_amdgcn_wmma_f32_16x16x32_bf16(false, qa0, false, kb2, (short)0, S1, false, false);
      S1 = __builtin_amdgcn_wmma_f32_16x16x32_bf16(false, qa1, false, kb3, (short)0, S1, false, false);
    }

    const float pj0 = position[j0 + cl];
    const float pj1 = position[j0 + 16 + cl];
    float cf[8];
#pragma unroll
    for (int r = 0; r < 8; r++) {
      const float rel0 = fminf(fabsf(posr[r] - pj0), 5000.f) * 0.0002f;
      const float rel1 = fminf(fabsf(posr[r] - pj1), 5000.f) * 0.0002f;
      const float e0 = (S0[r] + qwv[r] * rel0) * scale;
      const float e1 = (S1[r] + qwv[r] * rel1) * scale;
      float bm = half16_max(fmaxf(e0, e1));
      const float mn = fmaxf(mrow[r], bm);
      const float c  = __expf(mrow[r] - mn);
      const float p0 = __expf(e0 - mn);
      const float p1 = __expf(e1 - mn);
      lrow[r] = lrow[r] * c + half16_sum(p0 + p1);
      srow[r] = srow[r] * c + half16_sum(p0 * rel0 + p1 * rel1);
      mrow[r] = mn; cf[r] = c;
      const int prow = r + half * 8;
      pl[prow * 32 + cl]      = (bf16_t)p0;   // wave-private LDS tile: C-layout -> A-layout remap
      pl[prow * 32 + 16 + cl] = (bf16_t)p1;
    }
#pragma unroll
    for (int t = 0; t < 4; t++)
#pragma unroll
      for (int r = 0; r < 8; r++) O[t][r] *= cf[r];

    const v16bf pf = load_frag_lds(pl);       // P as 16x32 A fragment
#pragma unroll
    for (int t = 0; t < 4; t++) {
      v16bf vb = load_frag_nt(vpT, h * D_DIM + t * 16, L_DIM, j0);  // V^T rows contiguous
      O[t] = __builtin_amdgcn_wmma_f32_16x16x32_bf16(false, pf, false, vb, (short)0, O[t], false, false);
    }
  }

  float invl[8], sn[8];
#pragma unroll
  for (int r = 0; r < 8; r++) { invl[r] = 1.f / lrow[r]; sn[r] = srow[r] * invl[r]; }
#pragma unroll
  for (int t = 0; t < 4; t++) {
    const int colg = h * D_DIM + t * 16 + cl;
    const float wv = wvrel[colg];
#pragma unroll
    for (int r = 0; r < 8; r++) {
      const int row = i0 + r + half * 8;
      const float v = O[t][r] * invl[r] + sn[r] * wv;
      ctx[(size_t)row * E_DIM + colg] = (bf16_t)v;
    }
  }
}

// ---------------------------------------------------------------------------
extern "C" void kernel_launch(void* const* d_in, const int* in_sizes, int n_in,
                              void* d_out, int out_size, void* d_ws, size_t ws_size,
                              hipStream_t stream) {
  const float* V     = (const float*)d_in[0];
  const float* K     = (const float*)d_in[1];
  const float* Q     = (const float*)d_in[2];
  const float* pos   = (const float*)d_in[3];
  const float* Wq    = (const float*)d_in[4];
  const float* Wk    = (const float*)d_in[5];
  const float* Wv    = (const float*)d_in[6];
  const float* Wkrel = (const float*)d_in[7];
  const float* Wvrel = (const float*)d_in[8];
  const float* Wkabs = (const float*)d_in[9];
  const float* Wvabs = (const float*)d_in[10];
  const float* Wout  = (const float*)d_in[11];
  const float* bout  = (const float*)d_in[12];
  float* out = (float*)d_out;

  char* ws = (char*)d_ws;
  size_t off = 0;
  auto alloc = [&](size_t bytes) -> char* {
    char* p = ws + off;
    off += (bytes + 255) & ~(size_t)255;
    return p;
  };
  const size_t LE = (size_t)L_DIM * E_DIM;
  const size_t EE = (size_t)E_DIM * E_DIM;

  bf16_t* Qb   = (bf16_t*)alloc(LE * 2);
  bf16_t* Kb   = (bf16_t*)alloc(LE * 2);
  bf16_t* Vb   = (bf16_t*)alloc(LE * 2);
  bf16_t* Wqb  = (bf16_t*)alloc(EE * 2);
  bf16_t* Wkb  = (bf16_t*)alloc(EE * 2);
  bf16_t* Wvb  = (bf16_t*)alloc(EE * 2);
  bf16_t* Wob  = (bf16_t*)alloc(EE * 2);
  bf16_t* qp   = (bf16_t*)alloc(LE * 2);
  bf16_t* kp   = (bf16_t*)alloc(LE * 2);
  bf16_t* vpT  = (bf16_t*)alloc(LE * 2);   // transposed: [E][L]
  bf16_t* ctxb = (bf16_t*)alloc(LE * 2);
  float*  qwb  = (float*)alloc((size_t)H_DIM * L_DIM * 4);

  const dim3 cb(256);
  cvt_kernel<<<(int)(LE / 4 / 256), cb, 0, stream>>>(Q, Qb, (int)(LE / 4));
  cvt_kernel<<<(int)(LE / 4 / 256), cb, 0, stream>>>(K, Kb, (int)(LE / 4));
  cvt_kernel<<<(int)(LE / 4 / 256), cb, 0, stream>>>(V, Vb, (int)(LE / 4));
  cvt_kernel<<<(int)(EE / 4 / 256), cb, 0, stream>>>(Wq, Wqb, (int)(EE / 4));
  cvt_kernel<<<(int)(EE / 4 / 256), cb, 0, stream>>>(Wk, Wkb, (int)(EE / 4));
  cvt_kernel<<<(int)(EE / 4 / 256), cb, 0, stream>>>(Wv, Wvb, (int)(EE / 4));
  cvt_kernel<<<(int)(EE / 4 / 256), cb, 0, stream>>>(Wout, Wob, (int)(EE / 4));

  const dim3 gg(E_DIM / 128, L_DIM / 128), gb(256);
  // q = Q @ Wq^T                                  (bf16 out)
  gemm_nt_kernel<0, false, false><<<gg, gb, 0, stream>>>(
      Qb, Wqb, L_DIM, E_DIM, E_DIM, nullptr, nullptr, nullptr, qp, nullptr);
  // k = K @ Wk^T + pos (x) W_k_abs                (bf16 out)
  gemm_nt_kernel<0, true, false><<<gg, gb, 0, stream>>>(
      Kb, Wkb, L_DIM, E_DIM, E_DIM, pos, Wkabs, nullptr, kp, nullptr);
  // v^T = (V @ Wv^T + pos (x) W_v_abs)^T          (bf16 transposed out)
  gemm_nt_kernel<1, true, false><<<gg, gb, 0, stream>>>(
      Vb, Wvb, L_DIM, E_DIM, E_DIM, pos, Wvabs, nullptr, vpT, nullptr);

  qw_kernel<<<(H_DIM * L_DIM) / 256, cb, 0, stream>>>(qp, Wkrel, qwb);

  flash_kernel<<<H_DIM * (L_DIM / 128), cb, 0, stream>>>(qp, kp, vpT, qwb, pos, Wvrel, ctxb);

  // out = ctx @ Wout^T + b_out                    (f32 out)
  gemm_nt_kernel<2, false, true><<<gg, gb, 0, stream>>>(
      ctxb, Wob, L_DIM, E_DIM, E_DIM, nullptr, nullptr, bout, nullptr, out);
}
